// DecoupledGCN_20675972563382
// MI455X (gfx1250) — matrix operand order repaired and checked by reading, not compiled, not run on hardware
//
#include <hip/hip_runtime.h>
#include <hip/hip_bf16.h>

typedef __attribute__((ext_vector_type(16))) __bf16 v16bf;
typedef __attribute__((ext_vector_type(8)))  __bf16 v8bf;
typedef __attribute__((ext_vector_type(8)))  float  v8f;

// ---------------------------------------------------------------------------
// Pre-pass: pack-convert f32 -> bf16 (RNE), two elements per thread.
// Memory-bound one-shot; removes all conversion VALU from the GEMM hot loop.
// ---------------------------------------------------------------------------
__global__ void f32_to_bf16_pk(const float* __restrict__ in,
                               unsigned* __restrict__ out, long long npk)
{
    long long i = blockIdx.x * (long long)blockDim.x + threadIdx.x;
    if (i >= npk) return;
    const float2 v = ((const float2*)in)[i];
    unsigned lo = __builtin_bit_cast(unsigned, v.x);
    unsigned hi = __builtin_bit_cast(unsigned, v.y);
    lo = (lo + 0x7FFFu + ((lo >> 16) & 1u)) >> 16;     // round-to-nearest-even
    hi = (hi + 0x7FFFu + ((hi >> 16) & 1u)) >> 16;
    out[i] = lo | (hi << 16);
}

// ---------------------------------------------------------------------------
// h[N,64] = x[N,128] @ W[64,128]^T + b[64]   (bf16 WMMA, f32 accumulate)
// One wave -> 16 rows x 64 cols (4 C tiles x 4 K-steps = 16 WMMA).
// A/B fragments are straight b128 loads of pre-converted bf16.
// ---------------------------------------------------------------------------
__global__ __launch_bounds__(256) void gcn_gemm_wmma(
    const __bf16* __restrict__ xb, const __bf16* __restrict__ wb,
    const float* __restrict__ bias, float* __restrict__ h, int N)
{
    const int lane   = threadIdx.x & 31;
    const int wave   = threadIdx.x >> 5;
    const int tileM  = blockIdx.x * 8 + wave;
    const int tilesM = (N + 15) >> 4;
    if (tileM >= tilesM) return;            // wave-uniform: EXEC stays all-1s

    const int half = lane >> 4;             // 0 | 1
    const int l16  = lane & 15;

    int row = tileM * 16 + l16;
    if (row >= N) row = N - 1;              // clamp loads; stores are guarded
    const __bf16* __restrict__ xrow = xb + (size_t)row * 128;
    const int kA = half * 16;               // A: K = kA..kA+15
    const int kB = half * 16;               // B: K = kB..kB+15

    v8f acc[4];
    #pragma unroll
    for (int n = 0; n < 4; ++n) acc[n] = (v8f)(0.0f);

    #pragma unroll
    for (int s = 0; s < 4; ++s) {
        const int k0 = s * 32;
        const v8bf a_lo = *(const v8bf*)(xrow + k0 + kA);        // 16B
        const v8bf a_hi = *(const v8bf*)(xrow + k0 + kA + 8);    // 16B
        const v16bf a = __builtin_shufflevector(a_lo, a_hi,
                0, 1, 2, 3, 4, 5, 6, 7, 8, 9, 10, 11, 12, 13, 14, 15);
        #pragma unroll
        for (int n = 0; n < 4; ++n) {
            const v16bf bm = *(const v16bf*)(wb + (size_t)(n * 16 + l16) * 128 + k0 + kB);
            acc[n] = __builtin_amdgcn_wmma_f32_16x16x32_bf16(false, a, false, bm,
                                                             (short)0, acc[n], false, false);
        }
    }

    #pragma unroll
    for (int n = 0; n < 4; ++n) {
        const float bv = bias[n * 16 + l16];
        #pragma unroll
        for (int r = 0; r < 8; ++r) {
            const int m = tileM * 16 + r + 8 * half;  // C/D layout: M = r + 8*(lane>>4)
            if (m < N) h[(size_t)m * 64 + n * 16 + l16] = acc[n][r] + bv;
        }
    }
}

// ---------------------------------------------------------------------------
// Degree / hop kernels (L2-resident on MI455X: 25.6 MB matrix << 192 MB L2)
// ---------------------------------------------------------------------------
__global__ void deg_init(float* __restrict__ deg, int N) {
    int i = blockIdx.x * blockDim.x + threadIdx.x;
    if (i < N) deg[i] = 1.0f;                           // self loop
}

__global__ void deg_count(const int* __restrict__ src, float* __restrict__ deg, long long E) {
    long long e = blockIdx.x * (long long)blockDim.x + threadIdx.x;
    if (e < E) atomicAdd(&deg[src[e]], 1.0f);
}

__global__ void deg_rcp(float* __restrict__ deg, int N) {
    int i = blockIdx.x * blockDim.x + threadIdx.x;
    if (i < N) deg[i] = 1.0f / deg[i];                  // deg >= 1 always
}

__global__ void copy_f4(const float* __restrict__ in, float* __restrict__ out, long long n4) {
    long long i = blockIdx.x * (long long)blockDim.x + threadIdx.x;
    if (i < n4) ((float4*)out)[i] = ((const float4*)in)[i];
}

// 16 lanes per edge, float4 each: coalesced 256B gather + 4 f32 atomics/lane
__global__ void scatter_add(const float* __restrict__ hin, float* __restrict__ hout,
                            const int* __restrict__ src, const int* __restrict__ dst,
                            long long E)
{
    long long t = blockIdx.x * (long long)blockDim.x + threadIdx.x;
    long long e = t >> 4;
    if (e >= E) return;
    const int c = (int)(t & 15) * 4;
    const int s = src[e], d = dst[e];
    const float4 v = *(const float4*)(hin + (size_t)d * 64 + c);
    float* o = hout + (size_t)s * 64 + c;
    atomicAdd(o + 0, v.x); atomicAdd(o + 1, v.y);
    atomicAdd(o + 2, v.z); atomicAdd(o + 3, v.w);
}

__global__ void scale_rows(float* __restrict__ h, const float* __restrict__ rdeg, long long n4) {
    long long i = blockIdx.x * (long long)blockDim.x + threadIdx.x;
    if (i >= n4) return;
    const float r = rdeg[i >> 4];                       // 16 float4 per row of 64
    float4 v = ((float4*)h)[i];
    v.x *= r; v.y *= r; v.z *= r; v.w *= r;
    ((float4*)h)[i] = v;
}

// ---------------------------------------------------------------------------
extern "C" void kernel_launch(void* const* d_in, const int* in_sizes, int n_in,
                              void* d_out, int out_size, void* d_ws, size_t ws_size,
                              hipStream_t stream) {
    const float* x  = (const float*)d_in[0];
    const float* W  = (const float*)d_in[1];
    const float* b  = (const float*)d_in[2];
    const int*   ei = (const int*)d_in[3];

    const int OUT = in_sizes[2];                 // 64
    const int IN  = in_sizes[1] / OUT;           // 128
    const int N   = in_sizes[0] / IN;            // 100000
    const long long E = (long long)in_sizes[3] / 2;
    const int* src = ei;                         // scatter targets (edge_index[0])
    const int* dst = ei + E;                     // gather sources (edge_index[1])

    float* out  = (float*)d_out;
    float* bufA = (float*)d_ws;                          // h0: N*64 f32
    float* bufB = bufA + (size_t)N * 64;                 // h1: N*64 f32 (aliased as xb first)
    float* rdeg = bufB + (size_t)N * 64;                 // 1/deg: N f32
    __bf16* xb  = (__bf16*)bufB;                         // N*128 bf16 == N*64 f32 bytes
    __bf16* wb  = (__bf16*)(rdeg + N);                   // OUT*IN bf16 (16 KB)

    // 0) pack-convert x and W to bf16 (one-shot, memory bound)
    const long long npkX = (long long)N * IN / 2;
    const long long npkW = (long long)OUT * IN / 2;
    f32_to_bf16_pk<<<(unsigned)((npkX + 255) / 256), 256, 0, stream>>>(x, (unsigned*)xb, npkX);
    f32_to_bf16_pk<<<(unsigned)((npkW + 255) / 256), 256, 0, stream>>>(W, (unsigned*)wb, npkW);

    // 1) h0 = x @ W^T + b  (WMMA) ; consumes xb (bufB) before bufB is reused
    const int tilesM = (N + 15) / 16;
    gcn_gemm_wmma<<<(tilesM + 7) / 8, 256, 0, stream>>>(xb, wb, b, bufA, N);

    // 2) reciprocal degrees
    deg_init<<<(N + 255) / 256, 256, 0, stream>>>(rdeg, N);
    deg_count<<<(unsigned)((E + 255) / 256), 256, 0, stream>>>(src, rdeg, E);
    deg_rcp<<<(N + 255) / 256, 256, 0, stream>>>(rdeg, N);

    const long long n4 = (long long)N * (OUT / 4);          // float4 count
    const unsigned gN4 = (unsigned)((n4 + 255) / 256);
    const unsigned gE  = (unsigned)((E * 16 + 255) / 256);

    // 3) hop 1: bufB = (bufA + scatter(bufA)) * rdeg   (overwrites xb, no longer needed)
    copy_f4<<<gN4, 256, 0, stream>>>(bufA, bufB, n4);
    scatter_add<<<gE, 256, 0, stream>>>(bufA, bufB, src, dst, E);
    scale_rows<<<gN4, 256, 0, stream>>>(bufB, rdeg, n4);

    // 4) hop 2: out = (bufB + scatter(bufB)) * rdeg
    copy_f4<<<gN4, 256, 0, stream>>>(bufB, out, n4);
    scatter_add<<<gE, 256, 0, stream>>>(bufB, out, src, dst, E);
    scale_rows<<<gN4, 256, 0, stream>>>(out, rdeg, n4);
}
